// SimpleHarmonicRNN_56289841381378
// MI455X (gfx1250) — matrix operand order repaired and checked by reading, not compiled, not run on hardware
//
#include <hip/hip_runtime.h>
#include <hip/hip_bf16.h>

// ---------------------------------------------------------------------------
// SimpleHarmonicRNN for MI455X (gfx1250, wave32, WMMA, async-LDS)
//
// T=512, B=128, I=256, H=1024, O=256.  Recurrence independent per batch row
// -> persistent kernel, 8 WGs x 16 batch rows, t-loop with workgroup barriers.
// Latency hiding (serial phase):
//   * 32 waves/WG (8 per SIMD32), only 2 WMMA N-tiles per wave
//   * 4-stage software pipeline of B fragments (prefetch distance = 64 K),
//     schedule pinned with __builtin_amdgcn_sched_group_barrier
//   * global_load_async_to_lds_b128 prefetch of next step's U slab (ASYNCcnt)
// ---------------------------------------------------------------------------

#define T_STEPS 512
#define BATCH   128
#define IN_F    256
#define HID     1024
#define OUT_F   256
#define DT_C    0.01f
#define HSTR    1032   // bf16 LDS row stride (rows hit disjoint bank groups)
#define HVS     1036   // f32  LDS row stride (row +8 offset lands 32 banks away)

typedef __bf16 bf16;
typedef __attribute__((ext_vector_type(8)))  __bf16 bf16x8;
typedef __attribute__((ext_vector_type(16))) __bf16 bf16x16;
typedef __attribute__((ext_vector_type(8)))  float  f32x8;

__device__ __forceinline__ f32x8 wmma_bf16(bf16x16 a, bf16x16 b, f32x8 c) {
  return __builtin_amdgcn_wmma_f32_16x16x32_bf16(false, a, false, b, (short)0, c,
                                                 false, false);
}

// A operand (16x32 bf16): lane L -> row M = L&15; lanes 0-15: K=k..k+15;
// lanes 16-31: K=k+16..k+31.
__device__ __forceinline__ bf16x16 load_a_lds(const bf16* base, int m, int k, int hi) {
  union { bf16x16 v; bf16x8 h[2]; } u;
  const bf16* p = base + m * HSTR + k + hi * 16;
  u.h[0] = *(const bf16x8*)(p);
  u.h[1] = *(const bf16x8*)(p + 8);
  return u.v;
}

__device__ __forceinline__ bf16x16 load_a_gbl(const bf16* base, int ld, size_t row,
                                              int k, int hi) {
  union { bf16x16 v; bf16x8 h[2]; } u;
  const bf16* p = base + row * (size_t)ld + k + hi * 16;
  u.h[0] = *(const bf16x8*)(p);
  u.h[1] = *(const bf16x8*)(p + 8);
  return u.v;
}

// B operand (32x16 bf16) from row-major W[n][k] (computing A @ W^T):
// lane L -> weight row n0+(L&15), 16 contiguous K at k + (L>=16 ? 16 : 0).
__device__ __forceinline__ bf16x16 load_b_gbl(const bf16* W, int ld, int row,
                                              int kstart) {
  return *(const bf16x16*)(W + (size_t)row * ld + kstart);
}

// Pin the per-iteration pipeline: [B prefetch][A][WMMA][A][WMMA]
//                                 [B prefetch][A][WMMA][A][WMMA]
template <int NT>
__device__ __forceinline__ void sched_pipeline() {
  __builtin_amdgcn_sched_group_barrier(0x020, 4 * NT, 0);  // VMEM read: 2 stages
  __builtin_amdgcn_sched_group_barrier(0x100, 2, 0);       // DS read: A half
  __builtin_amdgcn_sched_group_barrier(0x008, NT, 0);      // WMMA
  __builtin_amdgcn_sched_group_barrier(0x100, 2, 0);
  __builtin_amdgcn_sched_group_barrier(0x008, NT, 0);
  __builtin_amdgcn_sched_group_barrier(0x020, 4 * NT, 0);
  __builtin_amdgcn_sched_group_barrier(0x100, 2, 0);
  __builtin_amdgcn_sched_group_barrier(0x008, NT, 0);
  __builtin_amdgcn_sched_group_barrier(0x100, 2, 0);
  __builtin_amdgcn_sched_group_barrier(0x008, NT, 0);
}

// K=HID tile-GEMM: C[NT tiles] += A(LDS) @ W^T.  4-stage rotating register
// pipeline of B fragments over 128-wide K blocks; branch-free body.  The
// final iteration prefetches 128 B past W's end -- dead values, and the
// workspace regions are contiguous so the reads stay in-bounds of d_ws.
template <int NT>
__device__ __forceinline__ void gemm_acc(f32x8* c, const bf16* aLds, const bf16* W,
                                         int tile0, int n, int hi) {
  bf16x16 s0[NT], s1[NT], s2[NT], s3[NT];
#pragma unroll
  for (int tt = 0; tt < NT; ++tt) {
    s0[tt] = load_b_gbl(W, HID, (tile0 + tt) * 16 + n, hi * 16);
    s1[tt] = load_b_gbl(W, HID, (tile0 + tt) * 16 + n, 32 + hi * 16);
  }
#pragma unroll 1
  for (int kk = 0; kk < HID; kk += 128) {
#pragma unroll
    for (int tt = 0; tt < NT; ++tt) {
      s2[tt] = load_b_gbl(W, HID, (tile0 + tt) * 16 + n, kk + 64 + hi * 16);
      s3[tt] = load_b_gbl(W, HID, (tile0 + tt) * 16 + n, kk + 96 + hi * 16);
    }
    bf16x16 a0 = load_a_lds(aLds, n, kk, hi);
#pragma unroll
    for (int tt = 0; tt < NT; ++tt) c[tt] = wmma_bf16(a0, s0[tt], c[tt]);
    bf16x16 a1 = load_a_lds(aLds, n, kk + 32, hi);
#pragma unroll
    for (int tt = 0; tt < NT; ++tt) c[tt] = wmma_bf16(a1, s1[tt], c[tt]);
#pragma unroll
    for (int tt = 0; tt < NT; ++tt) {
      s0[tt] = load_b_gbl(W, HID, (tile0 + tt) * 16 + n, kk + 128 + hi * 16);
      s1[tt] = load_b_gbl(W, HID, (tile0 + tt) * 16 + n, kk + 160 + hi * 16);
    }
    bf16x16 a2 = load_a_lds(aLds, n, kk + 64, hi);
#pragma unroll
    for (int tt = 0; tt < NT; ++tt) c[tt] = wmma_bf16(a2, s2[tt], c[tt]);
    bf16x16 a3 = load_a_lds(aLds, n, kk + 96, hi);
#pragma unroll
    for (int tt = 0; tt < NT; ++tt) c[tt] = wmma_bf16(a3, s3[tt], c[tt]);
    sched_pipeline<NT>();
  }
}

// Async global -> LDS copy of one 16x1024 bf16 slab into a padded [16][HSTR]
// LDS buffer.  1024 threads: thread = (row 0..15, seg 0..63), 2 x b128 each.
__device__ __forceinline__ void stage_u_async(const bf16* Ut, unsigned lds_base) {
  const int row = threadIdx.x >> 6;
  const int seg = threadIdx.x & 63;
  const char* g = (const char*)Ut + row * (HID * 2) + seg * 32;
  unsigned l = lds_base + row * (HSTR * 2) + seg * 32;
  asm volatile("global_load_async_to_lds_b128 %0, %1, off"
               :: "v"(l), "v"(g) : "memory");
  asm volatile("global_load_async_to_lds_b128 %0, %1, off"
               :: "v"(l + 16), "v"(g + 16) : "memory");
}

__device__ __forceinline__ void wait_async0() {
  asm volatile("s_wait_asynccnt 0x0" ::: "memory");
}

// ---------------------------------------------------------------------------
// K1: pack weights fp32 -> bf16 (split W_h into W_x | W_hh)
// ---------------------------------------------------------------------------
__global__ void hrnn_pack_weights(const float* __restrict__ W_h,
                                  const float* __restrict__ W_o,
                                  const float* __restrict__ W_out,
                                  bf16* __restrict__ Wx, bf16* __restrict__ Whh,
                                  bf16* __restrict__ Wo, bf16* __restrict__ Wou) {
  const size_t NWX  = (size_t)HID * IN_F;
  const size_t NWHH = (size_t)HID * HID;
  const size_t NWO  = (size_t)HID * HID;
  const size_t NWOU = (size_t)OUT_F * HID;
  const size_t TOT  = NWX + NWHH + NWO + NWOU;
  for (size_t i = (size_t)blockIdx.x * blockDim.x + threadIdx.x; i < TOT;
       i += (size_t)gridDim.x * blockDim.x) {
    if (i < NWX) {
      size_t h = i >> 8, k = i & 255;
      Wx[i] = (bf16)W_h[h * (IN_F + HID) + k];
    } else if (i < NWX + NWHH) {
      size_t j = i - NWX;
      size_t h = j >> 10, k = j & 1023;
      Whh[j] = (bf16)W_h[h * (IN_F + HID) + IN_F + k];
    } else if (i < NWX + NWHH + NWO) {
      size_t j = i - NWX - NWHH;
      Wo[j] = (bf16)W_o[j];
    } else {
      size_t j = i - NWX - NWHH - NWO;
      Wou[j] = (bf16)W_out[j];
    }
  }
}

// ---------------------------------------------------------------------------
// K2: convert x fp32 -> bf16
// ---------------------------------------------------------------------------
__global__ void hrnn_conv_x(const float* __restrict__ x, bf16* __restrict__ xb) {
  const size_t N = (size_t)T_STEPS * BATCH * IN_F;
  for (size_t i = (size_t)blockIdx.x * blockDim.x + threadIdx.x; i < N;
       i += (size_t)gridDim.x * blockDim.x)
    xb[i] = (bf16)x[i];
}

// ---------------------------------------------------------------------------
// K3: U[t*B+b, h] = sum_k x[t,b,k] W_x[h,k]   (time-parallel WMMA GEMM)
// 4096 blocks x 1024 threads; block = one 16-row M tile; wave owns 2 of the
// 64 N tiles; K = 256 as two branch-free 128-wide pipeline blocks.
// ---------------------------------------------------------------------------
__global__ __launch_bounds__(1024) void hrnn_input_gemm(const bf16* __restrict__ Xbf,
                                                        const bf16* __restrict__ Wx,
                                                        bf16* __restrict__ Ubf) {
  const int lane = threadIdx.x & 31;
  const int wave = threadIdx.x >> 5;   // 0..31
  const int hi   = lane >> 4;
  const int n    = lane & 15;
  const size_t row0 = (size_t)blockIdx.x * 16;
  const int r0 = (wave * 2) * 16 + n;      // weight row of tile 0
  const int r1 = (wave * 2 + 1) * 16 + n;  // weight row of tile 1

  f32x8 c[2] = {};
  bf16x16 s0[2], s1[2], s2[2], s3[2];
  s0[0] = load_b_gbl(Wx, IN_F, r0, hi * 16);
  s0[1] = load_b_gbl(Wx, IN_F, r1, hi * 16);
  s1[0] = load_b_gbl(Wx, IN_F, r0, 32 + hi * 16);
  s1[1] = load_b_gbl(Wx, IN_F, r1, 32 + hi * 16);
#pragma unroll 1
  for (int kk = 0; kk < IN_F; kk += 128) {
    s2[0] = load_b_gbl(Wx, IN_F, r0, kk + 64 + hi * 16);
    s2[1] = load_b_gbl(Wx, IN_F, r1, kk + 64 + hi * 16);
    s3[0] = load_b_gbl(Wx, IN_F, r0, kk + 96 + hi * 16);
    s3[1] = load_b_gbl(Wx, IN_F, r1, kk + 96 + hi * 16);
    bf16x16 a0 = load_a_gbl(Xbf, IN_F, row0 + n, kk, hi);
    c[0] = wmma_bf16(a0, s0[0], c[0]);
    c[1] = wmma_bf16(a0, s0[1], c[1]);
    bf16x16 a1 = load_a_gbl(Xbf, IN_F, row0 + n, kk + 32, hi);
    c[0] = wmma_bf16(a1, s1[0], c[0]);
    c[1] = wmma_bf16(a1, s1[1], c[1]);
    s0[0] = load_b_gbl(Wx, IN_F, r0, kk + 128 + hi * 16);
    s0[1] = load_b_gbl(Wx, IN_F, r1, kk + 128 + hi * 16);
    s1[0] = load_b_gbl(Wx, IN_F, r0, kk + 160 + hi * 16);
    s1[1] = load_b_gbl(Wx, IN_F, r1, kk + 160 + hi * 16);
    bf16x16 a2 = load_a_gbl(Xbf, IN_F, row0 + n, kk + 64, hi);
    c[0] = wmma_bf16(a2, s2[0], c[0]);
    c[1] = wmma_bf16(a2, s2[1], c[1]);
    bf16x16 a3 = load_a_gbl(Xbf, IN_F, row0 + n, kk + 96, hi);
    c[0] = wmma_bf16(a3, s3[0], c[0]);
    c[1] = wmma_bf16(a3, s3[1], c[1]);
  }
#pragma unroll
  for (int tt = 0; tt < 2; ++tt) {
    int h0 = (wave * 2 + tt) * 16;
#pragma unroll
    for (int r = 0; r < 8; ++r)
      Ubf[(row0 + r + hi * 8) * HID + h0 + n] = (bf16)c[tt][r];
  }
}

// ---------------------------------------------------------------------------
// K4: persistent recurrence.  8 blocks x 1024 threads (32 waves, 8/SIMD).
// Wave w owns h-tiles {2w, 2w+1} for GEMM1/GEMM2; waves 0-15 own one O-tile
// each for GEMM3.  LDS map (bytes):
//   hu_bf [16][HSTR] bf16 @ 0        ou_bf @ 33024
//   u0    @ 66048                    u1    @ 99072
//   hv f32 [16][HVS] @ 132096        ou f32 @ 198400     total 264704 B
// ---------------------------------------------------------------------------
__global__ __launch_bounds__(1024) void hrnn_recurrence(
    const bf16* __restrict__ Ubf, const bf16* __restrict__ Whh,
    const bf16* __restrict__ Wo, const bf16* __restrict__ Wou,
    const float* __restrict__ gain, const float* __restrict__ omega,
    const float* __restrict__ boff, const float* __restrict__ tau,
    const float* __restrict__ b_out, float* __restrict__ out) {
  extern __shared__ char smem[];
  bf16*  hu_bf = (bf16*)(smem);
  bf16*  ou_bf = (bf16*)(smem + 33024);
  bf16*  ubuf0 = (bf16*)(smem + 66048);
  bf16*  ubuf1 = (bf16*)(smem + 99072);
  float* hv_s  = (float*)(smem + 132096);
  float* ou_s  = (float*)(smem + 198400);
  const unsigned u0a = (unsigned)(uintptr_t)ubuf0;
  const unsigned u1a = (unsigned)(uintptr_t)ubuf1;

  const int lane = threadIdx.x & 31;
  const int wave = threadIdx.x >> 5;   // 0..31
  const int hi   = lane >> 4;
  const int n    = lane & 15;
  const int b0   = blockIdx.x * 16;

  // Per-lane coefficients for the 2 owned h columns.
  float g2[2], bc2[2], om22[2], al2[2];
#pragma unroll
  for (int tt = 0; tt < 2; ++tt) {
    int h = (wave * 2 + tt) * 16 + n;
    float om  = fabsf(omega[h]);
    float om2 = om * om;
    om22[tt] = om2;
    bc2[tt]  = om2 * 0.005f + fabsf(boff[h]);
    g2[tt]   = gain[h];
    al2[tt]  = __expf(-1.0f / fabsf(tau[h]));
  }

  // hu f32 state in registers (lane owns m = r+8*hi, h = tile*16+n).
  float hu[2][8];
#pragma unroll
  for (int tt = 0; tt < 2; ++tt)
#pragma unroll
    for (int r = 0; r < 8; ++r) hu[tt][r] = 0.f;
  f32x8 accv = {};

  for (int i = threadIdx.x; i < 16 * HSTR; i += 1024) {
    hu_bf[i] = (bf16)0.f;
    ou_bf[i] = (bf16)0.f;
  }
  for (int i = threadIdx.x; i < 16 * HVS; i += 1024) {
    hv_s[i] = 0.f;
    ou_s[i] = 0.f;
  }
  // Stage U slab for t = 0 into ubuf0.
  stage_u_async(Ubf + (size_t)b0 * HID, u0a);

  for (int t = 0; t < T_STEPS; ++t) {
    wait_async0();
    __syncthreads();               // U slab for t resident; prior step done

    const bf16* uL = (t & 1) ? ubuf1 : ubuf0;

    // ---- C init: in_sum accumulator starts at U_t (LDS-resident) ----
    f32x8 c[2];
#pragma unroll
    for (int tt = 0; tt < 2; ++tt) {
      int h0 = (wave * 2 + tt) * 16;
#pragma unroll
      for (int r = 0; r < 8; ++r)
        c[tt][r] = (float)uL[(r + hi * 8) * HSTR + h0 + n];
    }
    // Kick off async prefetch of the t+1 slab into the other buffer.
    if (t + 1 < T_STEPS)
      stage_u_async(Ubf + ((size_t)(t + 1) * BATCH + b0) * HID,
                    (t & 1) ? u0a : u1a);

    // ---- GEMM1: c += hu_old @ Whh^T ----
    gemm_acc<2>(c, hu_bf, Whh, wave * 2, n, hi);
    __syncthreads();               // all GEMM1 reads of hu_bf complete

    // ---- harmonic update (hv updated first, hu uses new hv) ----
#pragma unroll
    for (int tt = 0; tt < 2; ++tt) {
      int h0 = (wave * 2 + tt) * 16;
#pragma unroll
      for (int r = 0; r < 8; ++r) {
        int hv_i = (r + hi * 8) * HVS + h0 + n;
        float hvn = hv_s[hv_i] + hu[tt][r] * DT_C;
        float hun = hu[tt][r] + c[tt][r] * DT_C
                    - bc2[tt] * hu[tt][r] * (2.0f * DT_C)
                    - om22[tt] * hvn * DT_C;
        hun *= g2[tt];
        hv_s[hv_i] = hvn;
        hu[tt][r]  = hun;
        hu_bf[(r + hi * 8) * HSTR + h0 + n] = (bf16)hun;
      }
    }
    __syncthreads();               // hu_bf fully updated

    // ---- GEMM2: proj = hu_new @ Wo^T ; leaky integrate ----
#pragma unroll
    for (int tt = 0; tt < 2; ++tt) c[tt] = (f32x8){};
    gemm_acc<2>(c, hu_bf, Wo, wave * 2, n, hi);
#pragma unroll
    for (int tt = 0; tt < 2; ++tt) {
      int h0 = (wave * 2 + tt) * 16;
#pragma unroll
      for (int r = 0; r < 8; ++r) {
        int oi = (r + hi * 8) * HVS + h0 + n;
        float on = al2[tt] * ou_s[oi] + (1.0f - al2[tt]) * c[tt][r];
        ou_s[oi] = on;
        ou_bf[(r + hi * 8) * HSTR + h0 + n] = (bf16)on;
      }
    }
    __syncthreads();               // ou_bf fully updated

    // ---- GEMM3: acc += ou @ Wout^T (waves 0-15, one O-tile each) ----
    if (wave < 16)
      gemm_acc<1>(&accv, ou_bf, Wou, wave, n, hi);
  }

  // ---- epilogue: out = acc + T * b_out ----
  if (wave < 16) {
    int o0 = wave * 16;
    float bb = (float)T_STEPS * b_out[o0 + n];
#pragma unroll
    for (int r = 0; r < 8; ++r)
      out[(size_t)(b0 + r + hi * 8) * OUT_F + o0 + n] = accv[r] + bb;
  }
}

// ---------------------------------------------------------------------------
extern "C" void kernel_launch(void* const* d_in, const int* in_sizes, int n_in,
                              void* d_out, int out_size, void* d_ws, size_t ws_size,
                              hipStream_t stream) {
  const float* x     = (const float*)d_in[0];
  const float* W_h   = (const float*)d_in[1];
  const float* gain  = (const float*)d_in[2];
  const float* omega = (const float*)d_in[3];
  const float* boff  = (const float*)d_in[4];
  const float* W_o   = (const float*)d_in[5];
  const float* tau   = (const float*)d_in[6];
  const float* W_out = (const float*)d_in[7];
  const float* b_out = (const float*)d_in[8];
  float* out = (float*)d_out;

  char* ws = (char*)d_ws;
  bf16* xbf = (bf16*)(ws);                  //  33,554,432 B
  bf16* Wx  = (bf16*)(ws + 33554432);       //     524,288 B
  bf16* Whh = (bf16*)(ws + 34078720);       //   2,097,152 B
  bf16* Wo  = (bf16*)(ws + 36175872);       //   2,097,152 B
  bf16* Wou = (bf16*)(ws + 38273024);       //     524,288 B
  bf16* Ubf = (bf16*)(ws + 38797312);       // 134,217,728 B  (total ~165 MB)

  hrnn_pack_weights<<<1024, 256, 0, stream>>>(W_h, W_o, W_out, Wx, Whh, Wo, Wou);
  hrnn_conv_x<<<4096, 256, 0, stream>>>(x, xbf);
  hrnn_input_gemm<<<4096, 1024, 0, stream>>>(xbf, Wx, Ubf);

  size_t lds_bytes = 264704;  // see LDS map above
  hrnn_recurrence<<<8, 1024, lds_bytes, stream>>>(Ubf, Whh, Wo, Wou, gain, omega,
                                                  boff, tau, b_out, out);
}
